// CayleyConvolution_76699525972255
// MI455X (gfx1250) — matrix operand order repaired and checked by reading, not compiled, not run on hardware
//
#include <hip/hip_runtime.h>

typedef __attribute__((ext_vector_type(16))) _Float16 v16h;
typedef __attribute__((ext_vector_type(8)))  _Float16 v8h;
typedef __attribute__((ext_vector_type(4)))  _Float16 v4h;
typedef __attribute__((ext_vector_type(2)))  _Float16 v2h;
typedef __attribute__((ext_vector_type(8)))  float    v8f;

#define N_NODES 3072
#define F_IN    512
#define F_OUTD  256

#define BM 128
#define BN 128
#define BK 64
#define LDS_STR 72   // padded row stride (halfs) -> 144B; conflict-free b128 frag loads

// ---------------------------------------------------------------------------
// GEMM: C[M,N] = alpha * A[M,K] @ B[K,N] + beta * C
// A,B f16 (row-major), C f32. v_wmma_f32_16x16x32_f16, f32 accumulation.
// Software-pipelined: next tile's global loads issue before current compute.
// Requires M%128==0, N%128==0, K%64==0 (true for all shapes used here).
// ---------------------------------------------------------------------------
__global__ __launch_bounds__(256)
void gemm_f16_wmma(const _Float16* __restrict__ A, const _Float16* __restrict__ B,
                   float* __restrict__ C, int M, int N, int K,
                   float alpha, float beta) {
  __shared__ _Float16 As[BM * LDS_STR];
  __shared__ _Float16 Bs[BN * LDS_STR];

  const int tid  = threadIdx.x;
  const int lane = tid & 31;
  const int wave = tid >> 5;
  const int wm   = wave & 3;    // 0..3 (32 rows)
  const int wn   = wave >> 2;   // 0..1 (64 cols)
  const int m0   = blockIdx.y * BM;
  const int n0   = blockIdx.x * BN;
  const int l15     = lane & 15;
  const int halfsel = (lane >> 4) & 1;

  v8f acc[2][4];
#pragma unroll
  for (int i = 0; i < 2; ++i)
#pragma unroll
    for (int j = 0; j < 4; ++j)
      acc[i][j] = (v8f){0.f, 0.f, 0.f, 0.f, 0.f, 0.f, 0.f, 0.f};

  v8h aR[4];       // staged A: 4 x 8 halfs per thread (128x64 tile)
  v8h bR[2][2];    // staged B: 2 units x 2 k-rows x 8 halfs (64x128 tile)

  auto loadG = [&](int k0) {
#pragma unroll
    for (int i = 0; i < 4; ++i) {
      int li = tid + i * 256;          // 0..1023 octet units
      int r  = li >> 3;                // row 0..127
      int c8 = li & 7;                 // octet 0..7
      aR[i] = *(const v8h*)(A + (size_t)(m0 + r) * K + k0 + c8 * 8);
    }
#pragma unroll
    for (int i = 0; i < 2; ++i) {
      int u  = tid + i * 256;          // 0..511 units (2k x 8n)
      int kp = u & 31;                 // k-pair -> k = 2*kp
      int n8 = u >> 5;                 // col octet -> n = 8*n8
      bR[i][0] = *(const v8h*)(B + (size_t)(k0 + 2 * kp + 0) * N + n0 + n8 * 8);
      bR[i][1] = *(const v8h*)(B + (size_t)(k0 + 2 * kp + 1) * N + n0 + n8 * 8);
    }
  };

  auto storeL = [&]() {
#pragma unroll
    for (int i = 0; i < 4; ++i) {
      int li = tid + i * 256;
      int r  = li >> 3;
      int c8 = li & 7;
      v4h lo = __builtin_shufflevector(aR[i], aR[i], 0, 1, 2, 3);
      v4h hi = __builtin_shufflevector(aR[i], aR[i], 4, 5, 6, 7);
      *(v4h*)(&As[r * LDS_STR + c8 * 8 + 0]) = lo;
      *(v4h*)(&As[r * LDS_STR + c8 * 8 + 4]) = hi;
    }
#pragma unroll
    for (int i = 0; i < 2; ++i) {
      int u  = tid + i * 256;
      int kp = u & 31;
      int n8 = u >> 5;
#pragma unroll
      for (int j = 0; j < 8; ++j) {    // transpose: Bs[n][k] pairs
        v2h p;
        p.x = bR[i][0][j];
        p.y = bR[i][1][j];
        *(v2h*)(&Bs[(n8 * 8 + j) * LDS_STR + 2 * kp]) = p;
      }
    }
  };

  loadG(0);
  for (int k0 = 0; k0 < K; k0 += BK) {
    storeL();
    __syncthreads();
    if (k0 + BK < K) loadG(k0 + BK);   // prefetch next tile under compute

#pragma unroll
    for (int ks = 0; ks < 2; ++ks) {
      v16h af[2];
      v16h bf[4];
      // A frag: lane<16 -> K {0..7,16..23}, lane>=16 -> {8..15,24..31}
#pragma unroll
      for (int mi = 0; mi < 2; ++mi) {
        int r  = wm * 32 + mi * 16 + l15;
        int kb = ks * 32 + halfsel * 8;
        v8h lo = *(const v8h*)(&As[r * LDS_STR + kb]);
        v8h hi = *(const v8h*)(&As[r * LDS_STR + kb + 16]);
        af[mi] = __builtin_shufflevector(lo, hi, 0,1,2,3,4,5,6,7,8,9,10,11,12,13,14,15);
      }
      // B frag: lane<16 -> K {0..7,16..23}, lane>=16 -> {8..15,24..31} (matches A)
#pragma unroll
      for (int ni = 0; ni < 4; ++ni) {
        int n  = wn * 64 + ni * 16 + l15;
        int kb = ks * 32 + halfsel * 8;
        v8h lo = *(const v8h*)(&Bs[n * LDS_STR + kb]);
        v8h hi = *(const v8h*)(&Bs[n * LDS_STR + kb + 16]);
        bf[ni] = __builtin_shufflevector(lo, hi, 0,1,2,3,4,5,6,7,8,9,10,11,12,13,14,15);
      }
#pragma unroll
      for (int mi = 0; mi < 2; ++mi)
#pragma unroll
        for (int ni = 0; ni < 4; ++ni)
          acc[mi][ni] = __builtin_amdgcn_wmma_f32_16x16x32_f16(
              false, af[mi], false, bf[ni], (short)0, acc[mi][ni], false, false);
    }
    __syncthreads();
  }

  // epilogue: C layout VGPR r -> M = r + 8*(lane>=16), lane&15 -> N
  const int row_off = halfsel * 8;
#pragma unroll
  for (int mi = 0; mi < 2; ++mi) {
#pragma unroll
    for (int ni = 0; ni < 4; ++ni) {
      int mbase = m0 + wm * 32 + mi * 16 + row_off;
      int nc    = n0 + wn * 64 + ni * 16 + l15;
#pragma unroll
      for (int r = 0; r < 8; ++r) {
        size_t idx = (size_t)(mbase + r) * N + nc;
        float v = alpha * acc[mi][ni][r];
        if (beta != 0.0f) v += beta * C[idx];
        C[idx] = v;
      }
    }
  }
}

// ---------------------------------------------------------------------------
// Elementwise / reduction kernels (dual-write f32 + f16 mirrors where needed)
// ---------------------------------------------------------------------------
__global__ void build_hL_kernel(const float* __restrict__ adj,
                                const float* __restrict__ hp,
                                float* __restrict__ hL,
                                _Float16* __restrict__ hLh) {
  int idx = blockIdx.x * 256 + threadIdx.x;             // exact grid over N*N
  int row = idx / N_NODES, col = idx - row * N_NODES;
  float h = hp[0];
  float v = h * (((row == col) ? 1.0f : 0.0f) - adj[idx]);
  hL[idx]  = v;
  hLh[idx] = (_Float16)v;
}

__global__ void rowsum_dinv_kernel(const float* __restrict__ hL,
                                   float* __restrict__ d_r, float* __restrict__ d_i) {
  __shared__ float red[256];
  int row = blockIdx.x;
  float s = 0.f;
  for (int j = threadIdx.x; j < N_NODES; j += 256) s += hL[(size_t)row * N_NODES + j];
  red[threadIdx.x] = s;
  __syncthreads();
  for (int off = 128; off > 0; off >>= 1) {
    if (threadIdx.x < off) red[threadIdx.x] += red[threadIdx.x + off];
    __syncthreads();
  }
  if (threadIdx.x == 0) {
    float ss  = red[0];
    float inv = 1.0f / (ss * ss + 1.0f);                // diag_inv = (s - i)/(s^2+1)
    d_r[row] = ss * inv;
    d_i[row] = -inv;
  }
}

__global__ void cvt_f16_kernel(const float* __restrict__ src,
                               _Float16* __restrict__ dst, int n) {
  int idx = blockIdx.x * 256 + threadIdx.x;
  if (idx < n) dst[idx] = (_Float16)src[idx];
}

__global__ void init_y0_kernel(const float* __restrict__ hL,
                               float* __restrict__ y_r, float* __restrict__ y_i,
                               _Float16* __restrict__ y_rh, _Float16* __restrict__ y_ih,
                               float* __restrict__ xj_i) {
  int idx = blockIdx.x * 256 + threadIdx.x;
  int row = idx / N_NODES, col = idx - row * N_NODES;
  float a = hL[idx];
  y_r[idx]  = a;                                        // xj_r == hL (aliased)
  y_rh[idx] = (_Float16)a;
  float v = (row == col) ? -1.0f : 0.0f;                // xj_i = -I
  y_i[idx]  = v;
  y_ih[idx] = (_Float16)v;
  xj_i[idx] = v;
}

__global__ void jacobi_kernel(const float* __restrict__ xj_r, const float* __restrict__ xj_i,
                              const float* __restrict__ t_r,  const float* __restrict__ t_i,
                              float* __restrict__ y_r, float* __restrict__ y_i,
                              _Float16* __restrict__ y_rh, _Float16* __restrict__ y_ih,
                              const float* __restrict__ d_r, const float* __restrict__ d_i) {
  int idx = blockIdx.x * 256 + threadIdx.x;
  int row = idx / N_NODES;
  float yr = y_r[idx], yi = y_i[idx];
  float rr = xj_r[idx] - t_r[idx] + yi;                 // Re(xj - (hL@y + i*y))
  float ri = xj_i[idx] - t_i[idx] - yr;                 // Im(...)
  float dr = d_r[row], di = d_i[row];
  float nr = yr + dr * rr - di * ri;
  float ni = yi + dr * ri + di * rr;
  y_r[idx]  = nr;  y_i[idx]  = ni;
  y_rh[idx] = (_Float16)nr;
  y_ih[idx] = (_Float16)ni;
}

__global__ void make_xj1_kernel(const float* __restrict__ t_r, const float* __restrict__ t_i,
                                float* __restrict__ y_r, float* __restrict__ y_i,
                                _Float16* __restrict__ y_rh, _Float16* __restrict__ y_ih,
                                float* __restrict__ xj_r, float* __restrict__ xj_i) {
  int idx = blockIdx.x * 256 + threadIdx.x;
  float pr = y_r[idx], pi = y_i[idx];                   // prev = y(order 0)
  float a = t_r[idx] + pi;                              // xj_r = hL@prev_r + prev_i
  float b = t_i[idx] - pr;                              // xj_i = hL@prev_i - prev_r
  xj_r[idx] = a; xj_i[idx] = b;
  y_r[idx]  = a; y_i[idx]  = b;                         // y starts at xj
  y_rh[idx] = (_Float16)a;
  y_ih[idx] = (_Float16)b;
}

__global__ void relu_kernel(float* __restrict__ out, int n) {
  int idx = blockIdx.x * 256 + threadIdx.x;
  if (idx < n) out[idx] = fmaxf(out[idx], 0.0f);
}

// ---------------------------------------------------------------------------
// Host orchestration (graph-capture safe: only kernel launches on stream)
// ---------------------------------------------------------------------------
extern "C" void kernel_launch(void* const* d_in, const int* in_sizes, int n_in,
                              void* d_out, int out_size, void* d_ws, size_t ws_size,
                              hipStream_t stream) {
  const float* x   = (const float*)d_in[0];   // [3072,512]
  const float* adj = (const float*)d_in[1];   // [3072,3072]
  const float* h   = (const float*)d_in[2];   // scalar (device)
  const float* w0  = (const float*)d_in[3];   // [512,256]
  const float* wr  = (const float*)d_in[4];   // [2,512,256]
  const float* wi  = (const float*)d_in[5];   // [2,512,256]
  float* out = (float*)d_out;                 // [3072,256]

  const int N = N_NODES;
  char* ws = (char*)d_ws;
  const size_t NNf = (size_t)N * N * sizeof(float);     // 37748736
  const size_t NNh = (size_t)N * N * sizeof(_Float16);  // 18874368

  float* hL   = (float*)(ws + 0 * NNf);
  float* yr   = (float*)(ws + 1 * NNf);
  float* yi   = (float*)(ws + 2 * NNf);
  float* xjr  = (float*)(ws + 3 * NNf);
  float* xji  = (float*)(ws + 4 * NNf);
  float* tr   = (float*)(ws + 5 * NNf);
  float* ti   = (float*)(ws + 6 * NNf);
  char*  p    = ws + 7 * NNf;
  _Float16* hLh = (_Float16*)(p);              p += NNh;
  _Float16* yrh = (_Float16*)(p);              p += NNh;
  _Float16* yih = (_Float16*)(p);              p += NNh;
  float*    prer = (float*)(p);                p += (size_t)N * F_OUTD * 4;
  float*    prei = (float*)(p);                p += (size_t)N * F_OUTD * 4;
  _Float16* prerh = (_Float16*)(p);            p += (size_t)N * F_OUTD * 2;
  _Float16* preih = (_Float16*)(p);            p += (size_t)N * F_OUTD * 2;
  _Float16* xh   = (_Float16*)(p);             p += (size_t)N * F_IN * 2;
  _Float16* w0h  = (_Float16*)(p);             p += (size_t)F_IN * F_OUTD * 2;
  _Float16* wrh  = (_Float16*)(p);             p += (size_t)2 * F_IN * F_OUTD * 2;
  _Float16* wih  = (_Float16*)(p);             p += (size_t)2 * F_IN * F_OUTD * 2;
  float* dr = (float*)(p);                     p += (size_t)N * 4;
  float* di = (float*)(p);

  const dim3 tb(256);
  const dim3 ew((unsigned)((size_t)N * N / 256));       // exact: 36864 blocks

  auto gemm = [&](const _Float16* A, const _Float16* B, float* C, int M, int Nn, int K,
                  float alpha, float beta) {
    dim3 grid(Nn / BN, M / BM);
    gemm_f16_wmma<<<grid, tb, 0, stream>>>(A, B, C, M, Nn, K, alpha, beta);
  };
  auto cvt = [&](const float* s, _Float16* d, int n) {
    cvt_f16_kernel<<<(n + 255) / 256, tb, 0, stream>>>(s, d, n);
  };

  build_hL_kernel<<<ew, tb, 0, stream>>>(adj, h, hL, hLh);
  rowsum_dinv_kernel<<<N, tb, 0, stream>>>(hL, dr, di);

  cvt(x,  xh,  N * F_IN);
  cvt(w0, w0h, F_IN * F_OUTD);
  cvt(wr, wrh, 2 * F_IN * F_OUTD);
  cvt(wi, wih, 2 * F_IN * F_OUTD);

  // out = x @ w0
  gemm(xh, w0h, out, N, F_OUTD, F_IN, 1.0f, 0.0f);

  // ---- order 0: xj_r = hL, xj_i = -I ----
  init_y0_kernel<<<ew, tb, 0, stream>>>(hL, yr, yi, yrh, yih, xji);
  for (int it = 0; it < 3; ++it) {
    gemm(hLh, yrh, tr, N, N, N, 1.0f, 0.0f);
    gemm(hLh, yih, ti, N, N, N, 1.0f, 0.0f);
    jacobi_kernel<<<ew, tb, 0, stream>>>(hL, xji, tr, ti, yr, yi, yrh, yih, dr, di);
  }
  gemm(xh, wrh, prer, N, F_OUTD, F_IN, 1.0f, 0.0f);
  gemm(xh, wih, prei, N, F_OUTD, F_IN, 1.0f, 0.0f);
  cvt(prer, prerh, N * F_OUTD);
  cvt(prei, preih, N * F_OUTD);
  gemm(yrh, prerh, out, N, F_OUTD, N,  2.0f, 1.0f);     // out += 2*Re(y @ pre)
  gemm(yih, preih, out, N, F_OUTD, N, -2.0f, 1.0f);

  // ---- order 1: xj = hL@prev - i*prev ----
  gemm(hLh, yrh, tr, N, N, N, 1.0f, 0.0f);
  gemm(hLh, yih, ti, N, N, N, 1.0f, 0.0f);
  make_xj1_kernel<<<ew, tb, 0, stream>>>(tr, ti, yr, yi, yrh, yih, xjr, xji);
  for (int it = 0; it < 3; ++it) {
    gemm(hLh, yrh, tr, N, N, N, 1.0f, 0.0f);
    gemm(hLh, yih, ti, N, N, N, 1.0f, 0.0f);
    jacobi_kernel<<<ew, tb, 0, stream>>>(xjr, xji, tr, ti, yr, yi, yrh, yih, dr, di);
  }
  gemm(xh, wrh + (size_t)F_IN * F_OUTD, prer, N, F_OUTD, F_IN, 1.0f, 0.0f);
  gemm(xh, wih + (size_t)F_IN * F_OUTD, prei, N, F_OUTD, F_IN, 1.0f, 0.0f);
  cvt(prer, prerh, N * F_OUTD);
  cvt(prei, preih, N * F_OUTD);
  gemm(yrh, prerh, out, N, F_OUTD, N,  2.0f, 1.0f);
  gemm(yih, preih, out, N, F_OUTD, N, -2.0f, 1.0f);

  relu_kernel<<<(out_size + 255) / 256, tb, 0, stream>>>(out, out_size);

  (void)in_sizes; (void)n_in; (void)ws_size;
}
